// MSRBlock_77816217469148
// MI455X (gfx1250) — matrix operand order repaired and checked by reading, not compiled, not run on hardware
//
#include <hip/hip_runtime.h>

// ---------- types ----------
typedef __bf16 bf16;
typedef __attribute__((ext_vector_type(16))) __bf16 v16bf;
typedef __attribute__((ext_vector_type(8)))  __bf16 v8bf;
typedef __attribute__((ext_vector_type(8)))  float  v8f;

#define S_LEN 2048
#define D_DIM 1024
#define H_NUM 16
#define DH    64
#define B_NUM 2
#define ROWS  (B_NUM * S_LEN)   // 4096

// ---------- WMMA helpers (CDNA5 gfx1250, wave32) ----------
__device__ __forceinline__ v8f wmma_bf16(v16bf a, v16bf b, v8f c) {
    // D = A(16x32 bf16) * B(32x16 bf16) + C(16x16 f32)
    return __builtin_amdgcn_wmma_f32_16x16x32_bf16(
        /*neg_a=*/false, a, /*neg_b=*/false, b,
        /*c_mod=*/(short)0, c, /*reuse_a=*/false, /*reuse_b=*/false);
}

// A fragment 16x32 bf16: lane m = lane&15, half = lane>>4
// VGPR chunks hold K = [16*half, 16*half+8) and [16*half+8, 16*half+16)
__device__ __forceinline__ v16bf load_a_frag(const bf16* base, int ld, int lane) {
    int m = lane & 15, h = lane >> 4;
    const bf16* p = base + (size_t)m * ld + 16 * h;
    v8bf c0 = *(const v8bf*)p;
    v8bf c1 = *(const v8bf*)(p + 8);
    return __builtin_shufflevector(c0, c1, 0,1,2,3,4,5,6,7,8,9,10,11,12,13,14,15);
}

// B fragment 32x16 bf16 from BT (row-major [n][k]): lane n = lane&15, half = lane>>4
// lane holds K = [16*half, 16*half+16) contiguous -> 32B load
__device__ __forceinline__ v16bf load_b_frag(const bf16* base, int ld, int lane) {
    int n = lane & 15, h = lane >> 4;
    return *(const v16bf*)(base + (size_t)n * ld + 16 * h);
}

// ---------- conversion kernels ----------
__global__ void cvt_f32_bf16(const float* __restrict__ src, bf16* __restrict__ dst, int n) {
    int i = blockIdx.x * blockDim.x + threadIdx.x;
    if (i < n) dst[i] = (bf16)src[i];
}

__global__ void cvt_transpose_bf16(const float* __restrict__ W, bf16* __restrict__ WT, int dim) {
    int n = blockIdx.x * 16 + threadIdx.x;
    int k = blockIdx.y * 16 + threadIdx.y;
    if (n < dim && k < dim) WT[(size_t)n * dim + k] = (bf16)W[(size_t)k * dim + n];
}

// ---------- QKV projection GEMM ----------
// A = xb [4096 x 1024] bf16 rm, BT = Wqkv [3072 x 1024] bf16 (n-major)
// raw layout: [qkv][b][h][s][u]  f32
__global__ void qkv_gemm(const bf16* __restrict__ A, const bf16* __restrict__ BT,
                         float* __restrict__ raw) {
    int lane = threadIdx.x & 31;
    int w  = threadIdx.x >> 5;
    int wr = w & 3, wc = w >> 2;
    int row0 = blockIdx.x * 128 + wr * 32;
    int col0 = blockIdx.y * 128 + wc * 64;

    v8f z = {0.f,0.f,0.f,0.f,0.f,0.f,0.f,0.f};
    v8f o[2][4];
#pragma unroll
    for (int r = 0; r < 2; ++r)
#pragma unroll
        for (int c = 0; c < 4; ++c) o[r][c] = z;

    for (int kk = 0; kk < D_DIM; kk += 32) {
        v16bf a0 = load_a_frag(A + (size_t)row0 * D_DIM + kk, D_DIM, lane);
        v16bf a1 = load_a_frag(A + (size_t)(row0 + 16) * D_DIM + kk, D_DIM, lane);
#pragma unroll
        for (int ct = 0; ct < 4; ++ct) {
            v16bf b = load_b_frag(BT + (size_t)(col0 + 16 * ct) * D_DIM + kk, D_DIM, lane);
            o[0][ct] = wmma_bf16(a0, b, o[0][ct]);
            o[1][ct] = wmma_bf16(a1, b, o[1][ct]);
        }
    }
    int nlo = lane & 15, half = lane >> 4;
#pragma unroll
    for (int rt = 0; rt < 2; ++rt)
#pragma unroll
        for (int ct = 0; ct < 4; ++ct)
#pragma unroll
            for (int e = 0; e < 8; ++e) {
                int r = row0 + rt * 16 + e + 8 * half;
                int n = col0 + ct * 16 + nlo;
                int qkv = n >> 10, hh = (n >> 6) & 15, u = n & 63;
                int bb = r >> 11, sidx = r & 2047;
                size_t di = ((((size_t)qkv * B_NUM + bb) * H_NUM + hh) * S_LEN + sidx) * DH + u;
                raw[di] = o[rt][ct][e];
            }
}

// ---------- xPos rotate + pack: raw f32 -> Qr/Kr bf16 [b][h][s][u], Vt bf16 [b][h][u][s] ----------
__global__ void rotate_pack(const float* __restrict__ raw, bf16* __restrict__ Qr,
                            bf16* __restrict__ Kr, bf16* __restrict__ Vt) {
    int tid = blockIdx.x * blockDim.x + threadIdx.x;     // 2^21 threads
    int p    = tid & 31;            // pair index, u = 2p, 2p+1
    int sidx = (tid >> 5) & 2047;
    int hh   = (tid >> 16) & 15;
    int bb   = (tid >> 20) & 1;

    size_t baseQ = (((size_t)0 * B_NUM + bb) * H_NUM + hh) * S_LEN * DH + (size_t)sidx * DH;
    size_t baseK = (((size_t)1 * B_NUM + bb) * H_NUM + hh) * S_LEN * DH + (size_t)sidx * DH;
    size_t baseV = (((size_t)2 * B_NUM + bb) * H_NUM + hh) * S_LEN * DH + (size_t)sidx * DH;

    // angle = 10^(-4 * (2p/64)) = exp(-4*ln10 * p/32)
    float ang = __expf(-9.210340372f * (float)p * (1.0f / 32.0f));
    float na  = (float)sidx * ang;
    float sn = __sinf(na), cs = __cosf(na);

    size_t ob = (((size_t)bb * H_NUM + hh) * S_LEN + sidx) * DH + 2 * p;
    float q0 = raw[baseQ + 2 * p], q1 = raw[baseQ + 2 * p + 1];
    Qr[ob]     = (bf16)(q0 * cs - q1 * sn);
    Qr[ob + 1] = (bf16)(q0 * sn + q1 * cs);
    float k0 = raw[baseK + 2 * p], k1 = raw[baseK + 2 * p + 1];
    Kr[ob]     = (bf16)(k0 * cs - k1 * sn);
    Kr[ob + 1] = (bf16)(k0 * sn + k1 * cs);
    float v0 = raw[baseV + 2 * p], v1 = raw[baseV + 2 * p + 1];
    size_t vb = (((size_t)bb * H_NUM + hh) * DH + 2 * p) * S_LEN + sidx;
    Vt[vb]          = (bf16)v0;
    Vt[vb + S_LEN]  = (bf16)v1;
}

// ---------- retention attention (flash-style causal tiling) ----------
// one wave per 16 query rows per (b,h). R output f32 [b][s][h*64+u]
__global__ void retention_attn(const bf16* __restrict__ Qr, const bf16* __restrict__ Kr,
                               const bf16* __restrict__ Vt, float* __restrict__ R) {
    __shared__ bf16 sS[16][32];   // decayed S strip, re-layout C->A via LDS
    int lane = threadIdx.x & 31;
    int i0 = blockIdx.x * 16;
    int hh = blockIdx.y;
    int bb = blockIdx.z;

    const bf16* Qh = Qr + ((size_t)bb * H_NUM + hh) * S_LEN * DH;
    const bf16* Kh = Kr + ((size_t)bb * H_NUM + hh) * S_LEN * DH;
    const bf16* Vh = Vt + ((size_t)bb * H_NUM + hh) * DH * S_LEN;

    float gam = 1.0f - exp2f(-5.0f - (float)hh);
    float lng = __logf(gam);

    v16bf qf0 = load_a_frag(Qh + (size_t)i0 * DH + 0,  DH, lane);
    v16bf qf1 = load_a_frag(Qh + (size_t)i0 * DH + 32, DH, lane);

    v8f z = {0.f,0.f,0.f,0.f,0.f,0.f,0.f,0.f};
    v8f o[4];
#pragma unroll
    for (int ct = 0; ct < 4; ++ct) o[ct] = z;

    int nlo = lane & 15, half = lane >> 4;

    for (int j0 = 0; j0 < i0 + 16; j0 += 32) {
        // ---- S strip [16 x 32] = Q * K^T, with decay + causal mask ----
#pragma unroll
        for (int t = 0; t < 2; ++t) {
            v8f c = z;
            v16bf kf0 = load_b_frag(Kh + (size_t)(j0 + 16 * t) * DH + 0,  DH, lane);
            v16bf kf1 = load_b_frag(Kh + (size_t)(j0 + 16 * t) * DH + 32, DH, lane);
            c = wmma_bf16(qf0, kf0, c);
            c = wmma_bf16(qf1, kf1, c);
#pragma unroll
            for (int e = 0; e < 8; ++e) {
                int m = e + 8 * half;
                int i = i0 + m;
                int j = j0 + 16 * t + nlo;
                float dv = (float)(i - j);
                float val = (i >= j) ? c[e] * __expf(dv * lng) : 0.0f;
                sS[m][16 * t + nlo] = (bf16)val;
            }
        }
        // ---- O += S * V^T  (A frag re-loaded from LDS in A-layout) ----
        v16bf af = load_a_frag((const bf16*)&sS[0][0], 32, lane);
#pragma unroll
        for (int ct = 0; ct < 4; ++ct) {
            v16bf vb = load_b_frag(Vh + (size_t)(16 * ct) * S_LEN + j0, S_LEN, lane);
            o[ct] = wmma_bf16(af, vb, o[ct]);
        }
    }
    // epilogue: R[b][i][h*64 + u]
#pragma unroll
    for (int ct = 0; ct < 4; ++ct)
#pragma unroll
        for (int e = 0; e < 8; ++e) {
            int m = e + 8 * half;
            size_t di = ((size_t)bb * S_LEN + i0 + m) * D_DIM + hh * DH + 16 * ct + nlo;
            R[di] = o[ct][e];
        }
}

// ---------- group norm over dh=64 per head ----------
__global__ void group_norm_k(const float* __restrict__ R, const float* __restrict__ gamma,
                             const float* __restrict__ beta, bf16* __restrict__ Rn) {
    int row = blockIdx.x;          // 0..4095 == (b,s)
    int t = threadIdx.x;           // 256
    int g = t >> 4, q = t & 15;    // group = head, 16 lanes * 4 elems = 64
    size_t base = (size_t)row * D_DIM + g * 64 + q * 4;
    float v0 = R[base], v1 = R[base + 1], v2 = R[base + 2], v3 = R[base + 3];
    float s  = v0 + v1 + v2 + v3;
    float ss = v0 * v0 + v1 * v1 + v2 * v2 + v3 * v3;
#pragma unroll
    for (int m = 1; m < 16; m <<= 1) {
        s  += __shfl_xor(s,  m, 16);
        ss += __shfl_xor(ss, m, 16);
    }
    float mean = s * (1.0f / 64.0f);
    float var  = ss * (1.0f / 64.0f) - mean * mean;
    float rs = rsqrtf(var + 1e-5f);
    int di = g * 64 + q * 4;
    Rn[base]     = (bf16)((v0 - mean) * rs * gamma[di]     + beta[di]);
    Rn[base + 1] = (bf16)((v1 - mean) * rs * gamma[di + 1] + beta[di + 1]);
    Rn[base + 2] = (bf16)((v2 - mean) * rs * gamma[di + 2] + beta[di + 2]);
    Rn[base + 3] = (bf16)((v3 - mean) * rs * gamma[di + 3] + beta[di + 3]);
}

// ---------- output projection GEMM: y = Rn @ WO ----------
__global__ void out_gemm(const bf16* __restrict__ A, const bf16* __restrict__ BT,
                         float* __restrict__ C) {
    int lane = threadIdx.x & 31;
    int w  = threadIdx.x >> 5;
    int wr = w & 3, wc = w >> 2;
    int row0 = blockIdx.x * 128 + wr * 32;
    int col0 = blockIdx.y * 128 + wc * 64;

    v8f z = {0.f,0.f,0.f,0.f,0.f,0.f,0.f,0.f};
    v8f o[2][4];
#pragma unroll
    for (int r = 0; r < 2; ++r)
#pragma unroll
        for (int c = 0; c < 4; ++c) o[r][c] = z;

    for (int kk = 0; kk < D_DIM; kk += 32) {
        v16bf a0 = load_a_frag(A + (size_t)row0 * D_DIM + kk, D_DIM, lane);
        v16bf a1 = load_a_frag(A + (size_t)(row0 + 16) * D_DIM + kk, D_DIM, lane);
#pragma unroll
        for (int ct = 0; ct < 4; ++ct) {
            v16bf b = load_b_frag(BT + (size_t)(col0 + 16 * ct) * D_DIM + kk, D_DIM, lane);
            o[0][ct] = wmma_bf16(a0, b, o[0][ct]);
            o[1][ct] = wmma_bf16(a1, b, o[1][ct]);
        }
    }
    int nlo = lane & 15, half = lane >> 4;
#pragma unroll
    for (int rt = 0; rt < 2; ++rt)
#pragma unroll
        for (int ct = 0; ct < 4; ++ct)
#pragma unroll
            for (int e = 0; e < 8; ++e) {
                int r = row0 + rt * 16 + e + 8 * half;
                int n = col0 + ct * 16 + nlo;
                C[(size_t)r * D_DIM + n] = o[rt][ct][e];
            }
}

// ---------- host launcher ----------
extern "C" void kernel_launch(void* const* d_in, const int* in_sizes, int n_in,
                              void* d_out, int out_size, void* d_ws, size_t ws_size,
                              hipStream_t stream) {
    const float* x     = (const float*)d_in[0];
    const float* WQ    = (const float*)d_in[1];
    const float* WK    = (const float*)d_in[2];
    const float* WV    = (const float*)d_in[3];
    const float* WO    = (const float*)d_in[4];
    const float* gamma = (const float*)d_in[5];
    const float* beta  = (const float*)d_in[6];
    float* y = (float*)d_out;

    // workspace carve-up (all 256B aligned)
    char* p = (char*)d_ws;
    bf16* xb   = (bf16*)p;                 p += (size_t)ROWS * D_DIM * 2;        // 8 MiB
    bf16* Wb   = (bf16*)p;                 p += (size_t)3 * D_DIM * D_DIM * 2;   // 6 MiB
    bf16* WOt  = (bf16*)p;                 p += (size_t)D_DIM * D_DIM * 2;       // 2 MiB
    bf16* Qr   = (bf16*)p;                 p += (size_t)ROWS * D_DIM * 2;        // 8 MiB
    bf16* Kr   = (bf16*)p;                 p += (size_t)ROWS * D_DIM * 2;        // 8 MiB
    bf16* Vt   = (bf16*)p;                 p += (size_t)ROWS * D_DIM * 2;        // 8 MiB
    bf16* Rn   = (bf16*)p;                 p += (size_t)ROWS * D_DIM * 2;        // 8 MiB
    float* raw = (float*)p;                // 50.3 MiB, later reused as R f32 (16.8 MiB)
    float* R   = raw;

    const int nX = ROWS * D_DIM;        // 4,194,304
    const int nW = D_DIM * D_DIM;       // 1,048,576

    // 1) convert activations + QKV weights to bf16 (W already [n][k] layout)
    cvt_f32_bf16<<<(nX + 255) / 256, 256, 0, stream>>>(x, xb, nX);
    cvt_f32_bf16<<<(nW + 255) / 256, 256, 0, stream>>>(WQ, Wb, nW);
    cvt_f32_bf16<<<(nW + 255) / 256, 256, 0, stream>>>(WK, Wb + (size_t)nW, nW);
    cvt_f32_bf16<<<(nW + 255) / 256, 256, 0, stream>>>(WV, Wb + (size_t)2 * nW, nW);
    // 2) WO transpose-convert -> [n][k]
    cvt_transpose_bf16<<<dim3(64, 64), dim3(16, 16), 0, stream>>>(WO, WOt, D_DIM);
    // 3) QKV projection (WMMA bf16)
    qkv_gemm<<<dim3(ROWS / 128, (3 * D_DIM) / 128), 256, 0, stream>>>(xb, Wb, raw);
    // 4) xPos rotation + V transpose
    rotate_pack<<<(2 * 1024 * 1024) / 256, 256, 0, stream>>>(raw, Qr, Kr, Vt);
    // 5) causal retention attention (WMMA bf16, flash-style)
    retention_attn<<<dim3(S_LEN / 16, H_NUM, B_NUM), 32, 0, stream>>>(Qr, Kr, Vt, R);
    // 6) group norm -> bf16
    group_norm_k<<<ROWS, 256, 0, stream>>>(R, gamma, beta, Rn);
    // 7) output projection (WMMA bf16) -> f32 output
    out_gemm<<<dim3(ROWS / 128, D_DIM / 128), 256, 0, stream>>>(Rn, WOt, y);
}
